// NoisyDense_55422257988117
// MI455X (gfx1250) — matrix operand order repaired and checked by reading, not compiled, not run on hardware
//
#include <hip/hip_runtime.h>
#include <stdint.h>

// ---------------------------------------------------------------------------
// NoisyDense: out = relu(X @ (Wmu + Wsig * outer(eps_in, eps_out)) + bias)
// Phase 1: build bf16 X and bf16 transposed noisy-W (both L2-resident, 64MB).
// Phase 2: bf16 WMMA GEMM, 128x256 block tile, 8 waves each owning a 64x64
//          tile (16 v_wmma_f32_16x16x32_bf16 per BK=32 step), double-buffered
//          LDS fed by GLOBAL_LOAD_ASYNC_TO_LDS_B128 (ASYNCcnt), one barrier
//          per K-step. Bias+ReLU fused in the epilogue.
// Workspace: 64 MB (32 MB xb + 32 MB wT).
// ---------------------------------------------------------------------------

typedef __bf16 bf16;
typedef __attribute__((ext_vector_type(16))) __bf16 v16bf;
typedef __attribute__((ext_vector_type(8)))  __bf16 v8bf;
typedef __attribute__((ext_vector_type(8)))  float  v8f;

constexpr int DIM = 4096;          // BATCH = IN_DIM = UNITS
constexpr int BM  = 128;
constexpr int BN  = 256;
constexpr int BK  = 32;
constexpr int LDT = BK + 8;        // padded LDS row stride (bf16), 80B (16B mult)
constexpr int NK  = DIM / BK;

// --- CDNA5 async global->LDS helpers (inline asm; ASYNCcnt-tracked) --------
__device__ __forceinline__ uint32_t lds_off(const void* p) {
    // Flat LDS aperture keeps the LDS byte offset in addr[31:0].
    return (uint32_t)(uintptr_t)p;
}
__device__ __forceinline__ void async_copy_b128(uint32_t lds, const bf16* g) {
    asm volatile("global_load_async_to_lds_b128 %0, %1, off"
                 :: "v"(lds), "v"((uint64_t)(uintptr_t)g) : "memory");
}
__device__ __forceinline__ void wait_async0() {
    asm volatile("s_wait_asynccnt 0x0" ::: "memory");
}

// ---------------------------------------------------------------------------
// Phase 1a: x (fp32) -> xb (bf16), 8 elems/thread
// ---------------------------------------------------------------------------
__global__ void cvt_x_kernel(const float* __restrict__ x, bf16* __restrict__ xb) {
    const int i = (blockIdx.x * blockDim.x + threadIdx.x) * 8;
    const float4 f0 = reinterpret_cast<const float4*>(x + i)[0];
    const float4 f1 = reinterpret_cast<const float4*>(x + i)[1];
    v8bf o;
    o[0] = (bf16)f0.x; o[1] = (bf16)f0.y; o[2] = (bf16)f0.z; o[3] = (bf16)f0.w;
    o[4] = (bf16)f1.x; o[5] = (bf16)f1.y; o[6] = (bf16)f1.z; o[7] = (bf16)f1.w;
    *reinterpret_cast<v8bf*>(xb + i) = o;
}

// ---------------------------------------------------------------------------
// Phase 1b: wT[n][k] = bf16( wmu[k][n] + wsig[k][n]*eps_in[k]*eps_out[n] )
// 64x64 tile transpose through LDS.
// ---------------------------------------------------------------------------
__global__ void make_wT_kernel(const float* __restrict__ wmu,
                               const float* __restrict__ wsig,
                               const float* __restrict__ eps_in,
                               const float* __restrict__ eps_out,
                               bf16* __restrict__ wT) {
    __shared__ bf16 tile[64][72];
    const int n0 = blockIdx.x * 64;
    const int k0 = blockIdx.y * 64;
    const int t  = threadIdx.x;

    #pragma unroll
    for (int i = 0; i < 16; ++i) {
        const int idx = i * 256 + t;
        const int r = idx >> 6;     // k within tile
        const int c = idx & 63;     // n within tile
        const size_t g = (size_t)(k0 + r) * DIM + (n0 + c);
        const float v = wmu[g] + wsig[g] * eps_in[k0 + r] * eps_out[n0 + c];
        tile[c][r] = (bf16)v;
    }
    __syncthreads();

    #pragma unroll
    for (int g = t; g < 512; g += 256) {
        const int r = g >> 3;
        const int c = (g & 7) * 8;
        const uint4 val = *reinterpret_cast<const uint4*>(&tile[r][c]);
        *reinterpret_cast<uint4*>(wT + (size_t)(n0 + r) * DIM + (k0 + c)) = val;
    }
}

// ---------------------------------------------------------------------------
// Phase 2: bf16 WMMA GEMM with async double-buffered LDS pipeline.
// ---------------------------------------------------------------------------
__global__ __launch_bounds__(256)
void noisy_gemm_kernel(const bf16* __restrict__ A,    // [M][K] bf16
                       const bf16* __restrict__ Bt,   // [N][K] bf16
                       const float* __restrict__ b_mu,
                       const float* __restrict__ b_sig,
                       const float* __restrict__ eps_out,
                       float* __restrict__ out) {
    __shared__ bf16 ldsA[2][BM * LDT];   // 2 x 10240 B
    __shared__ bf16 ldsB[2][BN * LDT];   // 2 x 20480 B

    const int t    = threadIdx.x;
    const int lane = t & 31;
    const int wid  = t >> 5;       // 0..7
    const int wm   = wid >> 2;     // 0..1 : 64-row wave slab
    const int wn   = wid & 3;      // 0..3 : 64-col wave slab
    const int half = lane >> 4;
    const int l16  = lane & 15;

    const int blockM = blockIdx.y * BM;
    const int blockN = blockIdx.x * BN;

    // Copy maps (16B groups): A tile 512 groups, B tile 1024 groups.
    const int ar0 = t >> 2,         ac0 = (t & 3) * 8;
    const int ar1 = (t + 256) >> 2, ac1 = ((t + 256) & 3) * 8;

    const bf16* gA0 = A  + (size_t)(blockM + ar0) * DIM + ac0;
    const bf16* gA1 = A  + (size_t)(blockM + ar1) * DIM + ac1;
    const bf16* gB[4];
    int br[4], bc[4];
    #pragma unroll
    for (int q = 0; q < 4; ++q) {
        const int g = t + q * 256;
        br[q] = g >> 2; bc[q] = (g & 3) * 8;
        gB[q] = Bt + (size_t)(blockN + br[q]) * DIM + bc[q];
    }

    auto issue_tile = [&](int buf, int k0) {
        async_copy_b128(lds_off(&ldsA[buf][ar0 * LDT + ac0]), gA0 + k0);
        async_copy_b128(lds_off(&ldsA[buf][ar1 * LDT + ac1]), gA1 + k0);
        #pragma unroll
        for (int q = 0; q < 4; ++q)
            async_copy_b128(lds_off(&ldsB[buf][br[q] * LDT + bc[q]]), gB[q] + k0);
    };

    v8f acc[4][4] = {};

    float bias[4];
    #pragma unroll
    for (int j = 0; j < 4; ++j) {
        const int col = blockN + wn * 64 + j * 16 + l16;
        bias[j] = b_mu[col] + b_sig[col] * eps_out[col];
    }

    // Prologue: fill buffer 0
    issue_tile(0, 0);
    wait_async0();
    __syncthreads();

    for (int k = 0; k < NK; ++k) {
        const int buf = k & 1;
        if (k + 1 < NK) issue_tile(buf ^ 1, (k + 1) * BK);

        const bf16* sA = ldsA[buf];
        const bf16* sB = ldsB[buf];

        // B fragments: lane = column N (l16), 16 contiguous K at half*16
        v16bf bfrag[4];
        #pragma unroll
        for (int j = 0; j < 4; ++j) {
            const bf16* p = &sB[(wn * 64 + j * 16 + l16) * LDT + half * 16];
            const v8bf lo = *reinterpret_cast<const v8bf*>(p);
            const v8bf hi = *reinterpret_cast<const v8bf*>(p + 8);
            #pragma unroll
            for (int e = 0; e < 8; ++e) { bfrag[j][e] = lo[e]; bfrag[j][8 + e] = hi[e]; }
        }

        // A fragments: lane = row M (l16), 16 contiguous K at half*16
        #pragma unroll
        for (int i = 0; i < 4; ++i) {
            const bf16* p = &sA[(wm * 64 + i * 16 + l16) * LDT + half * 16];
            const v8bf lo = *reinterpret_cast<const v8bf*>(p);
            const v8bf hi = *reinterpret_cast<const v8bf*>(p + 8);
            v16bf afrag;
            #pragma unroll
            for (int e = 0; e < 8; ++e) { afrag[e] = lo[e]; afrag[8 + e] = hi[e]; }
            #pragma unroll
            for (int j = 0; j < 4; ++j) {
                acc[i][j] = __builtin_amdgcn_wmma_f32_16x16x32_bf16(
                    false, afrag, false, bfrag[j], (short)0, acc[i][j],
                    false, false);
            }
        }

        wait_async0();      // our async loads into buf^1 are complete
        __syncthreads();    // everyone's are; buf^1 ready, buf free next round
    }

    // Epilogue: C-layout — VGPR v: M = v + 8*half, N = l16
    #pragma unroll
    for (int i = 0; i < 4; ++i) {
        #pragma unroll
        for (int j = 0; j < 4; ++j) {
            const int col = blockN + wn * 64 + j * 16 + l16;
            const int row = blockM + wm * 64 + i * 16 + half * 8;
            #pragma unroll
            for (int v = 0; v < 8; ++v) {
                float val = acc[i][j][v] + bias[j];
                out[(size_t)(row + v) * DIM + col] = val > 0.f ? val : 0.f;
            }
        }
    }
}

// ---------------------------------------------------------------------------
extern "C" void kernel_launch(void* const* d_in, const int* in_sizes, int n_in,
                              void* d_out, int out_size, void* d_ws, size_t ws_size,
                              hipStream_t stream) {
    (void)in_sizes; (void)n_in; (void)out_size; (void)ws_size;

    const float* x     = (const float*)d_in[0];
    const float* w_mu  = (const float*)d_in[1];
    const float* w_sig = (const float*)d_in[2];
    const float* b_mu  = (const float*)d_in[3];
    const float* b_sig = (const float*)d_in[4];
    const float* e_in  = (const float*)d_in[5];
    const float* e_out = (const float*)d_in[6];
    float* out = (float*)d_out;

    bf16* xb = (bf16*)d_ws;
    bf16* wT = (bf16*)((char*)d_ws + (size_t)DIM * DIM * sizeof(bf16));

    cvt_x_kernel<<<DIM * DIM / (256 * 8), 256, 0, stream>>>(x, xb);
    make_wT_kernel<<<dim3(DIM / 64, DIM / 64), 256, 0, stream>>>(w_mu, w_sig, e_in, e_out, wT);
    noisy_gemm_kernel<<<dim3(DIM / BN, DIM / BM), 256, 0, stream>>>(
        xb, wT, b_mu, b_sig, e_out, out);
}
